// MiniGPT_2516850835850
// MI455X (gfx1250) — matrix operand order repaired and checked by reading, not compiled, not run on hardware
//
#include <hip/hip_runtime.h>
#include <hip/hip_bf16.h>
#include <stdint.h>

typedef __attribute__((ext_vector_type(16))) __bf16 v16bf;
typedef __attribute__((ext_vector_type(8)))  float  v8f;

namespace {
constexpr int kV = 32000, kD = 768, kH = 12, kL = 4, kT = 1024, kB = 4;
constexpr int kDFF = 4 * kD;      // 3072
constexpr int kM   = kB * kT;     // 4096 token rows
constexpr float kEps = 1e-5f;
}

// ---------- helpers ----------
__device__ inline unsigned short f2bf(float f) {
  unsigned int u = __float_as_uint(f);
  u += 0x7FFFu + ((u >> 16) & 1u);     // round-to-nearest-even bf16
  return (unsigned short)(u >> 16);
}

union FragBF { v16bf v; unsigned short u[16]; uint4 q[2]; };

// load a fragment as two contiguous 16B runs (ds_load_b128 / global_load_b128)
__device__ inline void ld_frag(FragBF& f, const unsigned short* p0, const unsigned short* p1) {
  f.q[0] = *(const uint4*)p0;
  f.q[1] = *(const uint4*)p1;
}

__device__ inline v8f v8zero() {
  v8f z;
#pragma unroll
  for (int i = 0; i < 8; ++i) z[i] = 0.f;
  return z;
}

// ---------- weight convert: fp32 [K,N] -> bf16 WMMA-B tiled layout ----------
// tile (tn, tk) holds B[tk*32 .. +32, tn*16 .. +16] as 512 bf16 in per-lane order:
//   addr = ((tn*Kt + tk) << 9) + lane*16 + i
//   value = B[tk*32 + (lane>=16 ? 16+i : i)][tn*16 + (lane&15)]
__global__ __launch_bounds__(256) void cvt_wtile_kernel(const float* __restrict__ in,
                                                        unsigned short* __restrict__ out,
                                                        int K, int N) {
  int o = blockIdx.x * 256 + threadIdx.x;   // < K*N (exact multiple of 256)
  int Kt = K >> 5;
  int tile   = o >> 9;
  int within = o & 511;
  int lane = within >> 4, i = within & 15;
  int tn = tile / Kt, tk = tile - tn * Kt;
  int k = tk * 32 + ((lane >= 16) ? 16 + i : i);
  int n = tn * 16 + (lane & 15);
  out[o] = f2bf(in[(long long)k * N + n]);
}

// ---------- elementwise kernels ----------
__global__ __launch_bounds__(256) void embed_kernel(const int* __restrict__ idx,
                                                    const float* __restrict__ tok,
                                                    const float* __restrict__ pos,
                                                    float* __restrict__ x) {
  long long i = (long long)blockIdx.x * 256 + threadIdx.x;   // < kM*kD
  int d = (int)(i % kD);
  long long bt = i / kD;
  int t = (int)(bt % kT);
  x[i] = tok[(long long)idx[bt] * kD + d] + pos[(long long)t * kD + d];
}

// one block (256 threads) per row; fp32 in, bf16 out
__global__ __launch_bounds__(256) void ln_kernel(const float* __restrict__ x,
                                                 const float* __restrict__ g,
                                                 const float* __restrict__ b,
                                                 unsigned short* __restrict__ out) {
  __shared__ float red[256];
  int row = blockIdx.x;
  const float* xr = x + (long long)row * kD;
  float s = 0.f;
  for (int c = threadIdx.x; c < kD; c += 256) s += xr[c];
  red[threadIdx.x] = s; __syncthreads();
  for (int st = 128; st > 0; st >>= 1) {
    if ((int)threadIdx.x < st) red[threadIdx.x] += red[threadIdx.x + st];
    __syncthreads();
  }
  float mu = red[0] / kD;
  __syncthreads();
  float v = 0.f;
  for (int c = threadIdx.x; c < kD; c += 256) { float d0 = xr[c] - mu; v += d0 * d0; }
  red[threadIdx.x] = v; __syncthreads();
  for (int st = 128; st > 0; st >>= 1) {
    if ((int)threadIdx.x < st) red[threadIdx.x] += red[threadIdx.x + st];
    __syncthreads();
  }
  float rstd = rsqrtf(red[0] / kD + kEps);
  unsigned short* orow = out + (long long)row * kD;
  for (int c = threadIdx.x; c < kD; c += 256)
    orow[c] = f2bf((xr[c] - mu) * rstd * g[c] + b[c]);
}

// ---------- bf16 WMMA GEMM:  C[M,N] = A[M,K] @ B[K,N] (+bias)(+resid)(ReLU) ----------
// A row-major bf16, B in WMMA-B tiled layout; both staged through LDS.
// block = 256 threads (8 waves), tile 256x64, BK=32; each wave owns 32 rows x 64 cols
// (2 A fragments x 4 B fragments = 8 WMMAs per k-step; B tile loaded once per block).
// M%256==0, N%64==0, K%32==0.
__global__ __launch_bounds__(256)
void gemm_bf16_kernel(const unsigned short* __restrict__ A,
                      const unsigned short* __restrict__ Bt,
                      const float* __restrict__ bias,
                      const float* __restrict__ resid,
                      float* __restrict__ outF,
                      unsigned short* __restrict__ outB,
                      int N, int K, int relu) {
  __shared__ alignas(16) unsigned short As[256][32];  // 16 KB
  __shared__ alignas(16) unsigned short Bs[4 * 512];  // 4 KB, fragment-tiled
  int tid = threadIdx.x;
  int lane = tid & 31, wv = tid >> 5;
  int half16 = lane >> 4;
  int l15 = lane & 15;
  int m0 = blockIdx.y * 256;
  int n0 = blockIdx.x * 64;
  int Kt = K >> 5;
  int tn0 = n0 >> 4;

  // staging: thread owns A row (m0+tid), 32 halfwords; and 8 halfwords of one B tile
  const unsigned short* arow = A + (long long)(m0 + tid) * K;
  int bnt = tid >> 6;             // 0..3  (which n-subtile)
  int bwo = (tid & 63) * 8;       // offset within 512-halfword tile
  long long btile0 = ((long long)(tn0 + bnt) * Kt) << 9;

  v8f acc[2][4];
#pragma unroll
  for (int f = 0; f < 2; ++f)
#pragma unroll
    for (int t = 0; t < 4; ++t) acc[f][t] = v8zero();

  for (int tk = 0; tk < Kt; ++tk) {
    const uint4* asrc = (const uint4*)(arow + tk * 32);
    uint4 av0 = asrc[0], av1 = asrc[1], av2 = asrc[2], av3 = asrc[3];
    uint4 bv = *(const uint4*)(Bt + btile0 + ((long long)tk << 9) + bwo);
    *(uint4*)&As[tid][0]  = av0;
    *(uint4*)&As[tid][8]  = av1;
    *(uint4*)&As[tid][16] = av2;
    *(uint4*)&As[tid][24] = av3;
    *(uint4*)&Bs[bnt * 512 + bwo] = bv;
    __syncthreads();

    FragBF a0, a1;
    ld_frag(a0, &As[wv * 32 + l15][16 * half16],      &As[wv * 32 + l15][16 * half16 + 8]);
    ld_frag(a1, &As[wv * 32 + 16 + l15][16 * half16], &As[wv * 32 + 16 + l15][16 * half16 + 8]);
#pragma unroll
    for (int nt = 0; nt < 4; ++nt) {
      FragBF bb;
      ld_frag(bb, &Bs[nt * 512 + lane * 16], &Bs[nt * 512 + lane * 16 + 8]);
      acc[0][nt] = __builtin_amdgcn_wmma_f32_16x16x32_bf16(
          false, a0.v, false, bb.v, (short)0, acc[0][nt], false, false);
      acc[1][nt] = __builtin_amdgcn_wmma_f32_16x16x32_bf16(
          false, a1.v, false, bb.v, (short)0, acc[1][nt], false, false);
    }
    __syncthreads();
  }

#pragma unroll
  for (int f = 0; f < 2; ++f) {
    int mrow = m0 + wv * 32 + f * 16 + 8 * half16;   // + i
#pragma unroll
    for (int nt = 0; nt < 4; ++nt) {
      int col = n0 + nt * 16 + l15;
      float bv = bias ? bias[col] : 0.f;
#pragma unroll
      for (int i = 0; i < 8; ++i) {
        long long off = (long long)(mrow + i) * N + col;
        float v = acc[f][nt][i] + bv;
        if (resid) v += resid[off];
        if (relu)  v = fmaxf(v, 0.f);
        if (outF) outF[off] = v;
        if (outB) outB[off] = f2bf(v);
      }
    }
  }
}

// ---------- flash attention (causal, hd=64), bf16 in/out, fp32 online softmax ----------
// grid (T/64, H, B), 128 threads (4 waves); wave w owns 16 query rows.
__global__ __launch_bounds__(128)
void flash_attn_kernel(const unsigned short* __restrict__ Q,
                       const unsigned short* __restrict__ Kb,
                       const unsigned short* __restrict__ Vb,
                       unsigned short* __restrict__ O) {
  __shared__ alignas(16) unsigned short Qs[64][64];    // [qrow][feat]
  __shared__ alignas(16) unsigned short Ks[32][64];    // [key][feat]
  __shared__ alignas(16) unsigned short Vt[64][32];    // [feat][key] (transposed)
  __shared__ alignas(16) unsigned short Ps[4][16][32]; // per-wave P tile

  int tid = threadIdx.x, lane = tid & 31, wv = tid >> 5;
  int half16 = lane >> 4;
  int l15 = lane & 15;
  int blk = blockIdx.x, h = blockIdx.y, bb = blockIdx.z;
  long long rowbase = (long long)bb * kT;
  int hc = h * 64;

  // load Q block 64x64 (32 halfwords = 64B per thread, vectorized)
  {
    int r = tid >> 1, c = (tid & 1) * 32;
    const uint4* src = (const uint4*)(Q + (rowbase + blk * 64 + r) * kD + hc + c);
    uint4* dst = (uint4*)&Qs[r][c];
#pragma unroll
    for (int e = 0; e < 4; ++e) dst[e] = src[e];
  }

  float mrow[8], lrow[8];
  v8f o[4];
#pragma unroll
  for (int i = 0; i < 8; ++i) { mrow[i] = -1e30f; lrow[i] = 0.f; }
#pragma unroll
  for (int t = 0; t < 4; ++t) o[t] = v8zero();

  int nblk = 2 * blk + 2;   // 32-key blocks for causal coverage
  for (int j = 0; j < nblk; ++j) {
    __syncthreads();        // protect Ks/Vt (and Qs on first pass) from prior consumers
    {
      int r = tid >> 2, c = (tid & 3) * 16;           // 16 halfwords per thread
      long long g = (rowbase + j * 32 + r) * kD + hc + c;
      const uint4* ksrc = (const uint4*)(Kb + g);
      uint4* kdst = (uint4*)&Ks[r][c];
      kdst[0] = ksrc[0]; kdst[1] = ksrc[1];
      FragBF vtmp;
      ld_frag(vtmp, Vb + g, Vb + g + 8);
#pragma unroll
      for (int e = 0; e < 16; ++e) Vt[c + e][r] = vtmp.u[e];   // transpose into LDS
    }
    __syncthreads();

    // S(16x32) = Qw(16x64) @ K^T(64x32): two 16x16 n-halves, K accumulated over 2x32
    v8f s0 = v8zero(), s1 = v8zero();
#pragma unroll
    for (int kc = 0; kc < 2; ++kc) {
      FragBF a;
      ld_frag(a, &Qs[wv * 16 + l15][kc * 32 + 16 * half16],
                 &Qs[wv * 16 + l15][kc * 32 + 16 * half16 + 8]);
      FragBF b0, b1;
      ld_frag(b0, &Ks[l15][kc * 32 + 16 * half16],      &Ks[l15][kc * 32 + 16 * half16 + 8]);
      ld_frag(b1, &Ks[16 + l15][kc * 32 + 16 * half16], &Ks[16 + l15][kc * 32 + 16 * half16 + 8]);
      s0 = __builtin_amdgcn_wmma_f32_16x16x32_bf16(false, a.v, false, b0.v, (short)0, s0, false, false);
      s1 = __builtin_amdgcn_wmma_f32_16x16x32_bf16(false, a.v, false, b1.v, (short)0, s1, false, false);
    }

    // online softmax per row (a C-fragment row lives at fixed i across a 16-lane half)
    int qrow0 = blk * 64 + wv * 16 + 8 * half16;
    int key0 = j * 32 + l15;
    int key1 = j * 32 + 16 + l15;
#pragma unroll
    for (int i = 0; i < 8; ++i) {
      int qr = qrow0 + i;
      float v0 = (key0 > qr) ? -1e30f : s0[i] * 0.125f;
      float v1 = (key1 > qr) ? -1e30f : s1[i] * 0.125f;
      float mx = fmaxf(v0, v1);
#pragma unroll
      for (int d0 = 1; d0 < 16; d0 <<= 1) mx = fmaxf(mx, __shfl_xor(mx, d0, 32));
      float mnew  = fmaxf(mrow[i], mx);
      float alpha = __expf(mrow[i] - mnew);
      float e0 = __expf(v0 - mnew);
      float e1 = __expf(v1 - mnew);
      float sum = e0 + e1;
#pragma unroll
      for (int d0 = 1; d0 < 16; d0 <<= 1) sum += __shfl_xor(sum, d0, 32);
      lrow[i] = lrow[i] * alpha + sum;
      mrow[i] = mnew;
#pragma unroll
      for (int t = 0; t < 4; ++t) o[t][i] *= alpha;
      Ps[wv][i + 8 * half16][l15]      = f2bf(e0);
      Ps[wv][i + 8 * half16][16 + l15] = f2bf(e1);
    }

    // O(16x64) += P(16x32) @ V(32x64)  (Ps wave-private; same-wave LDS RAW is in-order)
    FragBF a;
    ld_frag(a, &Ps[wv][l15][16 * half16], &Ps[wv][l15][16 * half16 + 8]);
#pragma unroll
    for (int nt = 0; nt < 4; ++nt) {
      FragBF bb;
      ld_frag(bb, &Vt[nt * 16 + l15][16 * half16], &Vt[nt * 16 + l15][16 * half16 + 8]);
      o[nt] = __builtin_amdgcn_wmma_f32_16x16x32_bf16(false, a.v, false, bb.v, (short)0, o[nt], false, false);
    }
  }

  // write normalized output (bf16, feeds the proj GEMM)
#pragma unroll
  for (int nt = 0; nt < 4; ++nt) {
#pragma unroll
    for (int i = 0; i < 8; ++i) {
      long long r = rowbase + blk * 64 + wv * 16 + 8 * half16 + i;
      O[r * kD + hc + nt * 16 + l15] = f2bf(o[nt][i] / lrow[i]);
    }
  }
}

// ---------- host ----------
extern "C" void kernel_launch(void* const* d_in, const int* in_sizes, int n_in,
                              void* d_out, int out_size, void* d_ws, size_t ws_size,
                              hipStream_t stream) {
  (void)in_sizes; (void)n_in; (void)out_size; (void)ws_size;
  const int*   idx   = (const int*)  d_in[0];
  const float* tok   = (const float*)d_in[1];
  const float* pos   = (const float*)d_in[2];
  const float* Wq    = (const float*)d_in[3];
  const float* Wk    = (const float*)d_in[4];
  const float* Wv    = (const float*)d_in[5];
  const float* Wp    = (const float*)d_in[6];
  const float* bproj = (const float*)d_in[7];
  const float* ln1g  = (const float*)d_in[8];
  const float* ln1b  = (const float*)d_in[9];
  const float* ln2g  = (const float*)d_in[10];
  const float* ln2b  = (const float*)d_in[11];
  const float* W1    = (const float*)d_in[12];
  const float* b1    = (const float*)d_in[13];
  const float* W2    = (const float*)d_in[14];
  const float* b2    = (const float*)d_in[15];
  const float* lnfg  = (const float*)d_in[16];
  const float* lnfb  = (const float*)d_in[17];
  const float* Wlm   = (const float*)d_in[18];
  const float* blm   = (const float*)d_in[19];

  char* ws = (char*)d_ws;
  size_t off = 0;
  auto alloc = [&](size_t bytes) -> char* {
    char* p = ws + off;
    off += (bytes + 255) & ~(size_t)255;
    return p;
  };
  // bf16 weight copies (WMMA-B tiled layout)
  unsigned short* wq_bf  = (unsigned short*)alloc((size_t)kL * kD * kD * 2);
  unsigned short* wk_bf  = (unsigned short*)alloc((size_t)kL * kD * kD * 2);
  unsigned short* wv_bf  = (unsigned short*)alloc((size_t)kL * kD * kD * 2);
  unsigned short* wp_bf  = (unsigned short*)alloc((size_t)kL * kD * kD * 2);
  unsigned short* w1_bf  = (unsigned short*)alloc((size_t)kL * kD * kDFF * 2);
  unsigned short* w2_bf  = (unsigned short*)alloc((size_t)kL * kDFF * kD * 2);
  unsigned short* wlm_bf = (unsigned short*)alloc((size_t)kD * kV * 2);
  // activations
  float*          x    = (float*)         alloc((size_t)kM * kD * 4);
  unsigned short* xn   = (unsigned short*)alloc((size_t)kM * kD * 2);
  unsigned short* qb   = (unsigned short*)alloc((size_t)kM * kD * 2);
  unsigned short* kb   = (unsigned short*)alloc((size_t)kM * kD * 2);
  unsigned short* vb   = (unsigned short*)alloc((size_t)kM * kD * 2);
  unsigned short* attn = (unsigned short*)alloc((size_t)kM * kD * 2);
  unsigned short* h1   = (unsigned short*)alloc((size_t)kM * kDFF * 2);

  auto cvtw = [&](const float* src, unsigned short* dst, int K, int N) {
    long long n = (long long)K * N;
    cvt_wtile_kernel<<<dim3((unsigned)(n / 256)), dim3(256), 0, stream>>>(src, dst, K, N);
  };
  for (int l = 0; l < kL; ++l) {
    cvtw(Wq + (size_t)l * kD * kD,   wq_bf + (size_t)l * kD * kD,   kD, kD);
    cvtw(Wk + (size_t)l * kD * kD,   wk_bf + (size_t)l * kD * kD,   kD, kD);
    cvtw(Wv + (size_t)l * kD * kD,   wv_bf + (size_t)l * kD * kD,   kD, kD);
    cvtw(Wp + (size_t)l * kD * kD,   wp_bf + (size_t)l * kD * kD,   kD, kD);
    cvtw(W1 + (size_t)l * kD * kDFF, w1_bf + (size_t)l * kD * kDFF, kD, kDFF);
    cvtw(W2 + (size_t)l * kDFF * kD, w2_bf + (size_t)l * kDFF * kD, kDFF, kD);
  }
  cvtw(Wlm, wlm_bf, kD, kV);

  embed_kernel<<<dim3((unsigned)((long long)kM * kD / 256)), dim3(256), 0, stream>>>(idx, tok, pos, x);

  auto gemm = [&](const unsigned short* A, const unsigned short* Bm, const float* bias,
                  const float* resid, float* oF, unsigned short* oB, int N, int K, int relu) {
    gemm_bf16_kernel<<<dim3(N / 64, kM / 256), dim3(256), 0, stream>>>(
        A, Bm, bias, resid, oF, oB, N, K, relu);
  };

  for (int l = 0; l < kL; ++l) {
    // x -> ln1 -> xn (bf16)
    ln_kernel<<<dim3(kM), dim3(256), 0, stream>>>(x, ln1g + (size_t)l * kD, ln1b + (size_t)l * kD, xn);
    // Q,K,V
    gemm(xn, wq_bf + (size_t)l * kD * kD, nullptr, nullptr, nullptr, qb, kD, kD, 0);
    gemm(xn, wk_bf + (size_t)l * kD * kD, nullptr, nullptr, nullptr, kb, kD, kD, 0);
    gemm(xn, wv_bf + (size_t)l * kD * kD, nullptr, nullptr, nullptr, vb, kD, kD, 0);
    // causal flash attention
    flash_attn_kernel<<<dim3(kT / 64, kH, kB), dim3(128), 0, stream>>>(qb, kb, vb, attn);
    // proj + bias + residual -> x (fp32)
    gemm(attn, wp_bf + (size_t)l * kD * kD, bproj + (size_t)l * kD, x, x, nullptr, kD, kD, 0);
    // ln2 -> xn
    ln_kernel<<<dim3(kM), dim3(256), 0, stream>>>(x, ln2g + (size_t)l * kD, ln2b + (size_t)l * kD, xn);
    // MLP
    gemm(xn, w1_bf + (size_t)l * kD * kDFF, b1 + (size_t)l * kDFF, nullptr, nullptr, h1, kDFF, kD, 1);
    gemm(h1, w2_bf + (size_t)l * kDFF * kD, b2 + (size_t)l * kD, x, x, nullptr, kD, kDFF, 0);
  }

  // final LN + LM head -> d_out (fp32 logits [B*T, V])
  ln_kernel<<<dim3(kM), dim3(256), 0, stream>>>(x, lnfg, lnfb, xn);
  gemm(xn, wlm_bf, blm, nullptr, (float*)d_out, nullptr, kV, kD, 0);
}
